// SupConRegLoss_52484500357720
// MI455X (gfx1250) — matrix operand order repaired and checked by reading, not compiled, not run on hardware
//
#include <hip/hip_runtime.h>
#include <math.h>

typedef __attribute__((ext_vector_type(8)))  float  v8f;
typedef __attribute__((ext_vector_type(8)))  __bf16 v8bf;
typedef __attribute__((ext_vector_type(16))) __bf16 v16bf;

#define NROW 4096
#define DIM  512
#define TEMP_INV (1.0f / 0.07f)
#define SSTRIDE 4100      // floats; 8*4100 % 64 == 32 -> M and M+8 store halves hit different banks
#define ASTRIDE_BF 520    // bf16;  1040B row stride -> 16B aligned, bank = 4*lane (conflict-free)
// LDS: S(16*SSTRIDE f32) + Ash/Asl(2 * 16*ASTRIDE_BF bf16) + labels(4096 f32) + red(512 f32)
#define LOSS_LDS_BYTES (16 * SSTRIDE * 4 + 2 * 16 * ASTRIDE_BF * 2 + NROW * 4 + 512 * 4)

// -------------------- 1) inverse row norms --------------------
__global__ void k_invnorm(const float* __restrict__ f, float* __restrict__ inv_norm) {
  __shared__ float red[128];
  const int row = blockIdx.x;
  const float4 v = reinterpret_cast<const float4*>(f + (size_t)row * DIM)[threadIdx.x];
  red[threadIdx.x] = v.x * v.x + v.y * v.y + v.z * v.z + v.w * v.w;
  __syncthreads();
  for (int off = 64; off > 0; off >>= 1) {
    if (threadIdx.x < off) red[threadIdx.x] += red[threadIdx.x + off];
    __syncthreads();
  }
  if (threadIdx.x == 0) inv_norm[row] = 1.0f / sqrtf(red[0]);
}

// -------------------- 2) bitonic argsort of labels (1 workgroup) --------------------
__global__ void k_sort(const float* __restrict__ labels, float* __restrict__ ls,
                       int* __restrict__ p) {
  __shared__ float key[NROW];
  __shared__ int   idx[NROW];
  const int tid = threadIdx.x;  // 1024
  for (int i = tid; i < NROW; i += 1024) { key[i] = labels[i]; idx[i] = i; }
  __syncthreads();
  for (int k = 2; k <= NROW; k <<= 1) {
    for (int j = k >> 1; j > 0; j >>= 1) {
      for (int i = tid; i < NROW; i += 1024) {
        const int ixj = i ^ j;
        if (ixj > i) {
          const bool up = ((i & k) == 0);
          const float a = key[i], b = key[ixj];
          const int ia = idx[i], ib = idx[ixj];
          const bool sw = up ? (a > b || (a == b && ia > ib))
                             : (a < b || (a == b && ia < ib));
          if (sw) { key[i] = b; key[ixj] = a; idx[i] = ib; idx[ixj] = ia; }
        }
      }
      __syncthreads();
    }
  }
  for (int i = tid; i < NROW; i += 1024) { ls[i] = key[i]; p[i] = idx[i]; }
}

// -------------------- 3) gather + normalize + bf16 hi/lo split --------------------
__global__ void k_permute(const float* __restrict__ f, const float* __restrict__ inv_norm,
                          const int* __restrict__ p, __bf16* __restrict__ gh,
                          __bf16* __restrict__ gl) {
  const int m = blockIdx.x;
  const int src = p[m];
  const float s = inv_norm[src];
  float4 v = reinterpret_cast<const float4*>(f + (size_t)src * DIM)[threadIdx.x];
  float x[4] = {v.x * s, v.y * s, v.z * s, v.w * s};
  union { __bf16 b[4]; uint2 u; } ph, pl;
#pragma unroll
  for (int c = 0; c < 4; ++c) {
    const __bf16 h = (__bf16)x[c];
    ph.b[c] = h;
    pl.b[c] = (__bf16)(x[c] - (float)h);
  }
  const size_t base = (size_t)m * DIM + threadIdx.x * 4;
  *reinterpret_cast<uint2*>(gh + base) = ph.u;
  *reinterpret_cast<uint2*>(gl + base) = pl.u;
}

// -------------------- 4) fused bf16x3 WMMA GEMM + exp + scan + loss --------------------
__global__ __launch_bounds__(512, 1)
void k_loss(const __bf16* __restrict__ gh, const __bf16* __restrict__ gl,
            const float* __restrict__ ls_g, float* __restrict__ partials) {
  extern __shared__ float smem[];
  float*  S   = smem;                                   // 16 x SSTRIDE f32
  __bf16* Ash = (__bf16*)(S + 16 * SSTRIDE);            // 16 x ASTRIDE_BF
  __bf16* Asl = Ash + 16 * ASTRIDE_BF;                  // 16 x ASTRIDE_BF
  float*  lsl = (float*)(Asl + 16 * ASTRIDE_BF);        // 4096 sorted labels
  float*  red = lsl + NROW;                             // 512 reduction scratch

  const int tid  = threadIdx.x;
  const int lane = tid & 31;
  const int wave = tid >> 5;     // 16 waves
  const int r0   = blockIdx.x * 16;

  for (int i = tid; i < NROW; i += 512) lsl[i] = ls_g[i];
  // stage A hi/lo tiles as dwords (stride 260 dwords per row)
  {
    unsigned int* dh = (unsigned int*)Ash;
    unsigned int* dl = (unsigned int*)Asl;
    const unsigned int* sh = (const unsigned int*)gh;
    const unsigned int* sl = (const unsigned int*)gl;
    for (int i = tid; i < 16 * (DIM / 2); i += 512) {
      const int row = i >> 8, kd = i & 255;
      dh[row * (ASTRIDE_BF / 2) + kd] = sh[(size_t)(r0 + row) * (DIM / 2) + kd];
      dl[row * (ASTRIDE_BF / 2) + kd] = sl[(size_t)(r0 + row) * (DIM / 2) + kd];
    }
  }
  __syncthreads();

  // WMMA lane mapping (bf16 16x16x32; A and B both use contiguous k = 16*(lane/16) + s)
  const int mrow = lane & 15;            // A: M row / B,D: N col
  const int ka   = (lane >> 4) << 4;     // A K-run base: 0 or 16
  const int kb   = (lane >> 4) << 4;     // B K-run base: 0 or 16
  const int rsel = (lane >> 4) << 3;     // D rows: +0 or +8

  for (int tile = wave; tile < NROW / 16; tile += 16) {
    const int c0 = tile * 16;
    const __bf16* __restrict__ ahr = Ash + mrow * ASTRIDE_BF;
    const __bf16* __restrict__ alr = Asl + mrow * ASTRIDE_BF;
    const __bf16* __restrict__ bhr = gh + (size_t)(c0 + mrow) * DIM + kb;
    const __bf16* __restrict__ blr = gl + (size_t)(c0 + mrow) * DIM + kb;
    if (tile + 16 < NROW / 16) {   // prefetch next tile's B rows (global_prefetch_b8)
      __builtin_prefetch(bhr + 256 * DIM, 0, 1);
      __builtin_prefetch(blr + 256 * DIM, 0, 1);
    }
    v8f acc = {};
#pragma unroll 4
    for (int k0 = 0; k0 < DIM; k0 += 32) {
      const v8bf ah0 = *(const v8bf*)(ahr + k0 + ka);
      const v8bf ah1 = *(const v8bf*)(ahr + k0 + ka + 8);
      const v8bf al0 = *(const v8bf*)(alr + k0 + ka);
      const v8bf al1 = *(const v8bf*)(alr + k0 + ka + 8);
      const v16bf ah = __builtin_shufflevector(ah0, ah1, 0, 1, 2, 3, 4, 5, 6, 7,
                                               8, 9, 10, 11, 12, 13, 14, 15);
      const v16bf al = __builtin_shufflevector(al0, al1, 0, 1, 2, 3, 4, 5, 6, 7,
                                               8, 9, 10, 11, 12, 13, 14, 15);
      const v16bf bh = *(const v16bf*)(bhr + k0);   // 32B aligned contiguous K-run
      const v16bf bl = *(const v16bf*)(blr + k0);
      acc = __builtin_amdgcn_wmma_f32_16x16x32_bf16(false, ah, false, bh, (short)0, acc,
                                                    false, false);
      acc = __builtin_amdgcn_wmma_f32_16x16x32_bf16(false, ah, false, bl, (short)0, acc,
                                                    false, false);
      acc = __builtin_amdgcn_wmma_f32_16x16x32_bf16(false, al, false, bh, (short)0, acc,
                                                    false, false);
    }
#pragma unroll
    for (int v = 0; v < 8; ++v) {
      S[(v + rsel) * SSTRIDE + c0 + mrow] = __expf(acc[v] * TEMP_INV);
    }
  }
  __syncthreads();

  // inclusive prefix scan: one row per wave (wave32 shuffle scan)
  {
    float* Sr = S + wave * SSTRIDE;
    float run = 0.0f;
    for (int c = 0; c < NROW; c += 32) {
      float v = Sr[c + lane];
#pragma unroll
      for (int d = 1; d < 32; d <<= 1) {
        const float t = __shfl_up(v, d, 32);
        if (lane >= d) v += t;
      }
      v += run;
      Sr[c + lane] = v;
      run = __shfl(v, 31, 32);
    }
  }
  __syncthreads();

  // loss: denom_{r,j} = total_r - interval sum over sorted labels in [l_r - t, l_r + t]
  float acc = 0.0f;
  for (int row = 0; row < 16; ++row) {
    const float* __restrict__ C = S + row * SSTRIDE;
    const int r = r0 + row;
    const float lr = lsl[r];
    const float total = C[NROW - 1];
    for (int j = tid; j < NROW; j += 512) {
      const float t = fabsf(lr - lsl[j]);
      const float xl = lr - t, xh = lr + t;
      int lo = 0, hi = NROW;                       // lower_bound: first >= xl
      while (lo < hi) { const int m = (lo + hi) >> 1; if (lsl[m] < xl) lo = m + 1; else hi = m; }
      int a = lo, b = NROW;                        // upper_bound: first > xh
      while (a < b) { const int m = (a + b) >> 1; if (lsl[m] <= xh) a = m + 1; else b = m; }
      const int hiIdx = a - 1;
      const float inner = (hiIdx >= lo) ? (C[hiIdx] - (lo ? C[lo - 1] : 0.0f)) : 0.0f;
      const float denom = total - inner;
      if (denom != 0.0f && j != r) {
        const float s = C[j] - (j ? C[j - 1] : 0.0f);
        acc += s / fmaxf(denom, 1e-4f);
      }
    }
  }
  red[tid] = acc;
  __syncthreads();
  for (int off = 256; off > 0; off >>= 1) {
    if (tid < off) red[tid] += red[tid + off];
    __syncthreads();
  }
  if (tid == 0) partials[blockIdx.x] = red[0];
}

// -------------------- 5) deterministic final reduce --------------------
__global__ void k_reduce(const float* __restrict__ partials, float* __restrict__ out) {
  __shared__ float red[256];
  red[threadIdx.x] = partials[threadIdx.x];
  __syncthreads();
  for (int off = 128; off > 0; off >>= 1) {
    if (threadIdx.x < off) red[threadIdx.x] += red[threadIdx.x + off];
    __syncthreads();
  }
  if (threadIdx.x == 0)
    out[0] = red[0] * (1.0f / ((float)(NROW - 1) * (float)NROW));
}

extern "C" void kernel_launch(void* const* d_in, const int* in_sizes, int n_in,
                              void* d_out, int out_size, void* d_ws, size_t ws_size,
                              hipStream_t stream) {
  const float* features = (const float*)d_in[0];   // 4096 x 512 f32
  const float* labels   = (const float*)d_in[1];   // 4096 f32
  float* out = (float*)d_out;

  char* ws = (char*)d_ws;
  const size_t gbytes = (size_t)NROW * DIM * 2;            // 4 MB each
  __bf16* gh      = (__bf16*)(ws);
  __bf16* gl      = (__bf16*)(ws + gbytes);
  float* ls       = (float*)(ws + 2 * gbytes);             // 16 KB
  int*   p        = (int*)  (ws + 2 * gbytes + 16384);     // 16 KB
  float* inv_norm = (float*)(ws + 2 * gbytes + 32768);     // 16 KB
  float* partials = (float*)(ws + 2 * gbytes + 49152);     // 1 KB

  (void)hipFuncSetAttribute((const void*)k_loss,
                            hipFuncAttributeMaxDynamicSharedMemorySize,
                            LOSS_LDS_BYTES);

  k_invnorm<<<NROW, 128, 0, stream>>>(features, inv_norm);
  k_sort   <<<1, 1024, 0, stream>>>(labels, ls, p);
  k_permute<<<NROW, 128, 0, stream>>>(features, inv_norm, p, gh, gl);
  k_loss   <<<NROW / 16, 512, LOSS_LDS_BYTES, stream>>>(gh, gl, ls, partials);
  k_reduce <<<1, 256, 0, stream>>>(partials, out);
}
